// MASMDDI_67877663146454
// MI455X (gfx1250) — compile-verified, matrix operand order checked by reading, not run on hardware
//
#include <hip/hip_runtime.h>
#include <hip/hip_bf16.h>

#define HEADS     4
#define N_NODES   50000
#define N_EDGES   400000
#define N_GRAPHS  512
#define IN_F      70
#define HID       128
#define N_LAYERS  3
#define QKV       (HEADS * HID)   // 512

typedef __attribute__((ext_vector_type(16))) _Float16 v16h;
typedef __attribute__((ext_vector_type(8)))  float    v8f;

// ---------- ordered-uint encoding of f32 for atomicMax on floats ----------
__device__ __forceinline__ unsigned enc_f32(float f) {
  unsigned u = __float_as_uint(f);
  return (u & 0x80000000u) ? ~u : (u | 0x80000000u);
}
__device__ __forceinline__ float dec_f32(unsigned u) {
  return __uint_as_float((u & 0x80000000u) ? (u ^ 0x80000000u) : ~u);
}
// enc(-inf) == 0x007FFFFF
#define ENC_NEG_INF 0x007FFFFFu

// =====================================================================
// Tiled GEMM, C = A(f32)[M,K] * B(f32)[K,N] + bias, via v_wmma_f32_16x16x32_f16.
// Block = 128 threads (4 waves), tile = 64(M) x 64(N), K step 32.
// Requirements honored by all call sites: N is a multiple of 64 and the
// grid covers N exactly, so only M (ragged last block) and K (lin0: 70)
// need range handling.
// =====================================================================
__global__ __launch_bounds__(128) void gemm_f16wmma(
    const float* __restrict__ A, const float* __restrict__ B,
    const float* __restrict__ bias, float* __restrict__ C,
    int M, int N, int K)
{
  __shared__ _Float16 As[64 * 32];   // [m][k]
  __shared__ _Float16 Bs[32 * 64];   // [k][n]

  const int tid  = threadIdx.x;
  const int lane = tid & 31;
  const int wv   = tid >> 5;          // 0..3: wave's 16-wide N slice
  const int row0 = blockIdx.y * 64;
  const int col0 = blockIdx.x * 64;

  v8f acc[4] = { {}, {}, {}, {} };

  const int arow = tid >> 1;          // 0..63
  const int acol = (tid & 1) * 16;    // 0 / 16
  const int brow = tid >> 2;          // 0..31
  const int bcol = (tid & 3) * 16;    // 0/16/32/48
  const int gr   = row0 + arow;
  const bool k16aligned = ((K & 3) == 0);   // row strips are float4-able

  for (int kt = 0; kt < K; kt += 32) {
    // ---- stage A tile (f32 -> f16), 16 contiguous k per thread ----
    if (k16aligned && gr < M && (kt + acol + 16) <= K) {
      const float4* ap = (const float4*)(A + (size_t)gr * K + kt + acol);
      #pragma unroll
      for (int j4 = 0; j4 < 4; ++j4) {
        float4 v4 = ap[j4];
        _Float16* d = &As[arow * 32 + acol + j4 * 4];
        d[0] = (_Float16)v4.x; d[1] = (_Float16)v4.y;
        d[2] = (_Float16)v4.z; d[3] = (_Float16)v4.w;
      }
    } else {
      // clamp-address + select-zero: no exec-mask branching around loads
      const int cr = (gr < M) ? gr : (M - 1);
      #pragma unroll
      for (int j = 0; j < 16; ++j) {
        int gk = kt + acol + j;
        int ck = (gk < K) ? gk : (K - 1);
        float vv = A[(size_t)cr * K + ck];
        As[arow * 32 + acol + j] =
            (_Float16)((gr < M && gk < K) ? vv : 0.f);
      }
    }
    // ---- stage B tile (f32 -> f16), 16 contiguous n per thread ----
    {
      int gk0 = kt + brow;
      if (gk0 < K) {
        const float4* bp = (const float4*)(B + (size_t)gk0 * N + col0 + bcol);
        #pragma unroll
        for (int j4 = 0; j4 < 4; ++j4) {
          float4 v4 = bp[j4];
          _Float16* d = &Bs[brow * 64 + bcol + j4 * 4];
          d[0] = (_Float16)v4.x; d[1] = (_Float16)v4.y;
          d[2] = (_Float16)v4.z; d[3] = (_Float16)v4.w;
        }
      } else {
        #pragma unroll
        for (int j = 0; j < 16; ++j)
          Bs[brow * 64 + bcol + j] = (_Float16)0.f;
      }
    }
    // prefetch next K strip (gfx1250 global_prefetch path)
    if (kt + 32 < K) {
      __builtin_prefetch(&A[(size_t)((gr < M) ? gr : 0) * K + kt + 32 + acol], 0, 1);
      __builtin_prefetch(&B[(size_t)(kt + 32 + brow) * N + col0 + bcol], 0, 1);
    }
    __syncthreads();

    const int g  = lane >> 4;                 // lane half
    const int fn = wv * 16 + (lane & 15);     // this wave's N column in tile

    // ---- B fragment: lane holds column fn, K per 16-bit WMMA layout ----
    v16h bf;
    #pragma unroll
    for (int j = 0; j < 4; ++j) {
      int k0 = (g ? 8 : 0) + 2 * j;
      bf[2 * j]     = Bs[(k0    ) * 64 + fn];
      bf[2 * j + 1] = Bs[(k0 + 1) * 64 + fn];
      int k1 = (g ? 24 : 16) + 2 * j;
      bf[8 + 2 * j]     = Bs[(k1    ) * 64 + fn];
      bf[8 + 2 * j + 1] = Bs[(k1 + 1) * 64 + fn];
    }

    // ---- 4 stacked M sub-tiles, A fragments are contiguous (b128) ----
    #pragma unroll
    for (int mt = 0; mt < 4; ++mt) {
      const int fm = mt * 16 + (lane & 15);
      v16h af;
      #pragma unroll
      for (int j = 0; j < 4; ++j) {
        int k0 = (g ? 8 : 0) + 2 * j;
        af[2 * j]     = As[fm * 32 + k0];
        af[2 * j + 1] = As[fm * 32 + k0 + 1];
        int k1 = (g ? 24 : 16) + 2 * j;
        af[8 + 2 * j]     = As[fm * 32 + k1];
        af[8 + 2 * j + 1] = As[fm * 32 + k1 + 1];
      }
      acc[mt] = __builtin_amdgcn_wmma_f32_16x16x32_f16(
          false, af, false, bf, (short)0, acc[mt], false, false);
    }
    __syncthreads();
  }

  // ---- store C (f32 C/D layout: VGPR r -> M = r + 8*(lane>>4)) ----
  const int g = lane >> 4;
  const int n = col0 + wv * 16 + (lane & 15);   // always < N (N % 64 == 0)
  const float bv = (bias != nullptr) ? bias[n] : 0.f;
  #pragma unroll
  for (int mt = 0; mt < 4; ++mt) {
    #pragma unroll
    for (int r = 0; r < 8; ++r) {
      int m = row0 + mt * 16 + 8 * g + r;
      if (m < M) C[(size_t)m * N + n] = acc[mt][r] + bv;
    }
  }
}

// =====================================================================
// Utility kernels
// =====================================================================
__global__ void fill_f32(float* p, float v, long long n) {
  long long i = (long long)blockIdx.x * blockDim.x + threadIdx.x;
  if (i < n) p[i] = v;
}
__global__ void fill_u32(unsigned* p, unsigned v, long long n) {
  long long i = (long long)blockIdx.x * blockDim.x + threadIdx.x;
  if (i < n) p[i] = v;
}
__global__ void relu_copy(const float* __restrict__ in, float* __restrict__ out, long long n) {
  long long i = (long long)blockIdx.x * blockDim.x + threadIdx.x;
  if (i < n) out[i] = fmaxf(in[i], 0.f);
}
__global__ void pool_add(const float* __restrict__ h, const int* __restrict__ batch,
                         float* __restrict__ embed) {
  long long i = (long long)blockIdx.x * blockDim.x + threadIdx.x;
  if (i >= (long long)N_NODES * HID) return;
  int node = (int)(i >> 7);
  int d    = (int)(i & 127);
  atomicAdd(&embed[(size_t)batch[node] * HID + d], h[i]);
}

// =====================================================================
// Edge attention: pass A — alpha + running max per (dst, head)
// one wave per (edge, head)
// =====================================================================
__global__ __launch_bounds__(256) void attn_alpha(
    const float* __restrict__ q, const float* __restrict__ k,
    const int* __restrict__ src, const int* __restrict__ dst,
    float* __restrict__ alpha, unsigned* __restrict__ mmax)
{
  long long wid = ((long long)blockIdx.x * blockDim.x + threadIdx.x) >> 5;
  int lane = threadIdx.x & 31;
  if (wid >= (long long)N_EDGES * HEADS) return;
  int e = (int)(wid >> 2), h = (int)(wid & 3);
  int s = src[e], d = dst[e];
  const float* qp = q + (size_t)d * QKV + h * HID;
  const float* kp = k + (size_t)s * QKV + h * HID;
  float sum = 0.f;
  #pragma unroll
  for (int jj = 0; jj < 4; ++jj) sum += qp[lane + jj * 32] * kp[lane + jj * 32];
  #pragma unroll
  for (int off = 16; off; off >>= 1) sum += __shfl_xor(sum, off, 32);
  if (lane == 0) {
    float a = sum * 0.08838834764831845f;   // 1/sqrt(128)
    alpha[(size_t)e * HEADS + h] = a;
    atomicMax(&mmax[(size_t)d * HEADS + h], enc_f32(a));
  }
}

// pass B — exp(alpha - max), accumulate denominator. one thread per (e,h)
__global__ void attn_exp(float* __restrict__ alpha, const unsigned* __restrict__ mmax,
                         const int* __restrict__ dst, float* __restrict__ den)
{
  long long i = (long long)blockIdx.x * blockDim.x + threadIdx.x;
  if (i >= (long long)N_EDGES * HEADS) return;
  int e = (int)(i >> 2), h = (int)(i & 3);
  int d = dst[e];
  float m  = dec_f32(mmax[(size_t)d * HEADS + h]);
  float ev = __expf(alpha[i] - m);
  alpha[i] = ev;
  atomicAdd(&den[(size_t)d * HEADS + h], ev);
}

// pass C — out[dst] += (e/den) * v[src] / HEADS. one wave per (e,h).
// out was pre-initialized with the skip branch, so this fuses mean+skip.
__global__ __launch_bounds__(256) void attn_agg(
    const float* __restrict__ alpha, const float* __restrict__ den,
    const float* __restrict__ v, const int* __restrict__ src,
    const int* __restrict__ dst, float* __restrict__ out)
{
  long long wid = ((long long)blockIdx.x * blockDim.x + threadIdx.x) >> 5;
  int lane = threadIdx.x & 31;
  if (wid >= (long long)N_EDGES * HEADS) return;
  int e = (int)(wid >> 2), h = (int)(wid & 3);
  int s = src[e], d = dst[e];
  float coef = alpha[(size_t)e * HEADS + h] / den[(size_t)d * HEADS + h] * 0.25f;
  const float* vp = v + (size_t)s * QKV + h * HID;
  float* op = out + (size_t)d * HID;
  #pragma unroll
  for (int jj = 0; jj < 4; ++jj)
    atomicAdd(&op[lane + jj * 32], coef * vp[lane + jj * 32]);
}

// =====================================================================
// Tail: cross-attention interaction + relational bilinear score.
// One 128-thread block per graph.
// score_b = sum_i h_i * (row_i . t) / max(||row_i||, eps)
// =====================================================================
__global__ __launch_bounds__(128) void final_score(
    const float* __restrict__ he, const float* __restrict__ te,
    const float* __restrict__ qv, const float* __restrict__ kv,
    const int* __restrict__ rels, const float* __restrict__ rel_emb,
    float* __restrict__ score)
{
  __shared__ float sh[128], st[128], red[128];
  int b = blockIdx.x, t = threadIdx.x;
  size_t base = (size_t)b * HID + t;
  float inter = tanhf(qv[base] * kv[base]);
  float hv = he[base] + inter;
  float tv = te[base] + inter;

  red[t] = hv * hv; __syncthreads();
  for (int s2 = 64; s2 > 0; s2 >>= 1) { if (t < s2) red[t] += red[t + s2]; __syncthreads(); }
  float hn = fmaxf(sqrtf(red[0]), 1e-12f); __syncthreads();

  red[t] = tv * tv; __syncthreads();
  for (int s2 = 64; s2 > 0; s2 >>= 1) { if (t < s2) red[t] += red[t + s2]; __syncthreads(); }
  float tn = fmaxf(sqrtf(red[0]), 1e-12f); __syncthreads();

  sh[t] = hv / hn; st[t] = tv / tn; __syncthreads();

  const float* R = rel_emb + (size_t)rels[b] * (HID * HID);
  int lane = t & 31, w = t >> 5;
  float acc = 0.f;
  for (int i = w; i < HID; i += 4) {
    const float* row = R + (size_t)i * HID;
    float dot = 0.f, nn = 0.f;
    #pragma unroll
    for (int jj = 0; jj < 4; ++jj) {
      float r = row[lane + jj * 32];
      dot += r * st[lane + jj * 32];
      nn  += r * r;
    }
    #pragma unroll
    for (int off = 16; off; off >>= 1) {
      dot += __shfl_xor(dot, off, 32);
      nn  += __shfl_xor(nn,  off, 32);
    }
    acc += sh[i] * dot / fmaxf(sqrtf(nn), 1e-12f);
  }
  if (lane == 0) red[w] = acc;
  __syncthreads();
  if (t == 0) score[b] = red[0] + red[1] + red[2] + red[3];
}

// =====================================================================
// Host-side drivers
// =====================================================================
static inline void launch_gemm(const float* A, const float* B, const float* bias,
                               float* C, int M, int N, int K, hipStream_t s)
{
  dim3 grid((N + 63) / 64, (M + 63) / 64);
  gemm_f16wmma<<<grid, 128, 0, s>>>(A, B, bias, C, M, N, K);
}

struct EncBufs {
  float *h, *q, *k, *v, *alpha, *den, *out;
  unsigned* m;
};

static void encode(const float* x, const int* eidx, const int* batch,
                   const float* lin0_W, const float* lin0_b,
                   const float* Wq, const float* bq, const float* Wk, const float* bk,
                   const float* Wv, const float* bv, const float* Ws, const float* bs,
                   float* embed, const EncBufs& wsb, hipStream_t s)
{
  const int* src = eidx;
  const int* dst = eidx + N_EDGES;

  launch_gemm(x, lin0_W, lin0_b, wsb.h, N_NODES, HID, IN_F, s);

  const long long nEH   = (long long)N_EDGES * HEADS;      // (edge, head) pairs
  const int blocksWave  = (int)((nEH * 32 + 255) / 256);   // wave-per-pair kernels
  const int blocksThr   = (int)((nEH + 255) / 256);
  const long long nNH   = (long long)N_NODES * HEADS;
  const long long nH    = (long long)N_NODES * HID;

  for (int l = 0; l < N_LAYERS; ++l) {
    const float* Wql = Wq + (size_t)l * HID * QKV;
    const float* Wkl = Wk + (size_t)l * HID * QKV;
    const float* Wvl = Wv + (size_t)l * HID * QKV;
    const float* Wsl = Ws + (size_t)l * HID * HID;
    launch_gemm(wsb.h, Wql, bq + (size_t)l * QKV, wsb.q, N_NODES, QKV, HID, s);
    launch_gemm(wsb.h, Wkl, bk + (size_t)l * QKV, wsb.k, N_NODES, QKV, HID, s);
    launch_gemm(wsb.h, Wvl, bv + (size_t)l * QKV, wsb.v, N_NODES, QKV, HID, s);
    // skip branch pre-initializes the aggregation target
    launch_gemm(wsb.h, Wsl, bs + (size_t)l * HID, wsb.out, N_NODES, HID, HID, s);

    fill_u32<<<(unsigned)((nNH + 255) / 256), 256, 0, s>>>(wsb.m, ENC_NEG_INF, nNH);
    fill_f32<<<(unsigned)((nNH + 255) / 256), 256, 0, s>>>(wsb.den, 0.f, nNH);

    attn_alpha<<<blocksWave, 256, 0, s>>>(wsb.q, wsb.k, src, dst, wsb.alpha, wsb.m);
    attn_exp  <<<blocksThr,  256, 0, s>>>(wsb.alpha, wsb.m, dst, wsb.den);
    attn_agg  <<<blocksWave, 256, 0, s>>>(wsb.alpha, wsb.den, wsb.v, src, dst, wsb.out);

    relu_copy<<<(unsigned)((nH + 255) / 256), 256, 0, s>>>(wsb.out, wsb.h, nH);
  }

  const long long nE = (long long)N_GRAPHS * HID;
  fill_f32<<<(unsigned)((nE + 255) / 256), 256, 0, s>>>(embed, 0.f, nE);
  pool_add<<<(unsigned)((nH + 255) / 256), 256, 0, s>>>(wsb.h, batch, embed);
}

extern "C" void kernel_launch(void* const* d_in, const int* in_sizes, int n_in,
                              void* d_out, int out_size, void* d_ws, size_t ws_size,
                              hipStream_t stream)
{
  (void)in_sizes; (void)n_in; (void)out_size; (void)ws_size;

  const float* h_x     = (const float*)d_in[0];
  const int*   h_ei    = (const int*)  d_in[1];
  const int*   h_batch = (const int*)  d_in[2];
  const float* t_x     = (const float*)d_in[3];
  const int*   t_ei    = (const int*)  d_in[4];
  const int*   t_batch = (const int*)  d_in[5];
  const int*   rels    = (const int*)  d_in[6];
  const float* lin0_W  = (const float*)d_in[7];
  const float* lin0_b  = (const float*)d_in[8];
  const float* Wq      = (const float*)d_in[9];
  const float* bq      = (const float*)d_in[10];
  const float* Wk      = (const float*)d_in[11];
  const float* bk      = (const float*)d_in[12];
  const float* Wv      = (const float*)d_in[13];
  const float* bv      = (const float*)d_in[14];
  const float* Ws      = (const float*)d_in[15];
  const float* bs      = (const float*)d_in[16];
  const float* wq_W    = (const float*)d_in[17];
  const float* wq_b    = (const float*)d_in[18];
  const float* wk_W    = (const float*)d_in[19];
  const float* wk_b    = (const float*)d_in[20];
  const float* rel_emb = (const float*)d_in[21];

  // ---- carve workspace ----
  char* p = (char*)d_ws;
  auto carve = [&](size_t bytes) -> void* {
    void* r = (void*)p;
    p += (bytes + 255) & ~(size_t)255;
    return r;
  };
  float*    b_h     = (float*)   carve(sizeof(float) * (size_t)N_NODES * HID);
  float*    b_q     = (float*)   carve(sizeof(float) * (size_t)N_NODES * QKV);
  float*    b_k     = (float*)   carve(sizeof(float) * (size_t)N_NODES * QKV);
  float*    b_v     = (float*)   carve(sizeof(float) * (size_t)N_NODES * QKV);
  float*    b_alpha = (float*)   carve(sizeof(float) * (size_t)N_EDGES * HEADS);
  float*    b_den   = (float*)   carve(sizeof(float) * (size_t)N_NODES * HEADS);
  float*    b_out   = (float*)   carve(sizeof(float) * (size_t)N_NODES * HID);
  unsigned* b_m     = (unsigned*)carve(sizeof(unsigned) * (size_t)N_NODES * HEADS);
  float*    e_h     = (float*)   carve(sizeof(float) * (size_t)N_GRAPHS * HID);
  float*    e_t     = (float*)   carve(sizeof(float) * (size_t)N_GRAPHS * HID);
  float*    c_q     = (float*)   carve(sizeof(float) * (size_t)N_GRAPHS * HID);
  float*    c_k     = (float*)   carve(sizeof(float) * (size_t)N_GRAPHS * HID);

  EncBufs wsb { b_h, b_q, b_k, b_v, b_alpha, b_den, b_out, b_m };

  encode(h_x, h_ei, h_batch, lin0_W, lin0_b, Wq, bq, Wk, bk, Wv, bv, Ws, bs,
         e_h, wsb, stream);
  encode(t_x, t_ei, t_batch, lin0_W, lin0_b, Wq, bq, Wk, bk, Wv, bv, Ws, bs,
         e_t, wsb, stream);

  launch_gemm(e_h, wq_W, wq_b, c_q, N_GRAPHS, HID, HID, stream);
  launch_gemm(e_t, wk_W, wk_b, c_k, N_GRAPHS, HID, HID, stream);

  final_score<<<N_GRAPHS, 128, 0, stream>>>(e_h, e_t, c_q, c_k, rels, rel_emb,
                                            (float*)d_out);
}
